// W8A8Linear_68813966017108
// MI455X (gfx1250) — matrix operand order repaired and checked by reading, not compiled, not run on hardware
//
#include <hip/hip_runtime.h>
#include <hip/hip_bf16.h>

typedef _Float16 v16h __attribute__((ext_vector_type(16)));
typedef _Float16 v8h  __attribute__((ext_vector_type(8)));
typedef float    v8f  __attribute__((ext_vector_type(8)));

union H16 { v16h v; v8h h[2]; };

// ---------------------------------------------------------------------------
// Kernel 1: 2:4 activation sparsity + per-token absmax fake-quant.
// ---------------------------------------------------------------------------
__global__ __launch_bounds__(256) void quant_sparsify_kernel(
    const float* __restrict__ x,
    const float* __restrict__ sscale,
    _Float16* __restrict__ qx,
    float* __restrict__ ts,
    int K)
{
    const int token = blockIdx.x;
    const int tid   = threadIdx.x;
    const int base  = tid * 16;

    const float* xr = x + (size_t)token * K;

    float vals[16];
    float mx = 0.0f;

    #pragma unroll
    for (int g = 0; g < 4; ++g) {
        float v[4], met[4];
        #pragma unroll
        for (int j = 0; j < 4; ++j) {
            int f = base + g * 4 + j;
            v[j]   = xr[f];
            met[j] = fabsf(v[j]) * sscale[f];
        }
        #pragma unroll
        for (int j = 0; j < 4; ++j) {
            int rank = 0;                       // stable ascending rank
            #pragma unroll
            for (int i = 0; i < 4; ++i)
                rank += (met[i] < met[j]) || (met[i] == met[j] && i < j);
            float keep = (rank >= 2) ? v[j] : 0.0f;   // zero the 2 smallest
            vals[g * 4 + j] = keep;
            mx = fmaxf(mx, fabsf(keep));
        }
    }

    __shared__ float red[256];
    red[tid] = mx;
    __syncthreads();
    for (int s = 128; s > 0; s >>= 1) {
        if (tid < s) red[tid] = fmaxf(red[tid], red[tid + s]);
        __syncthreads();
    }
    const float scale = fmaxf(red[0], 1e-5f) / 127.0f;

    _Float16* qr = qx + (size_t)token * K + base;
    #pragma unroll
    for (int j = 0; j < 16; ++j)
        qr[j] = (_Float16)__builtin_rintf(vals[j] / scale);   // half-to-even
    if (tid == 0) ts[token] = scale;
}

// ---------------------------------------------------------------------------
// Kernel 2: split fp32 weight into f16 hi + f16 residual lo.
// ---------------------------------------------------------------------------
__global__ __launch_bounds__(256) void weight_split_kernel(
    const float* __restrict__ w,
    _Float16* __restrict__ whi,
    _Float16* __restrict__ wlo,
    long long total)
{
    long long i = (long long)blockIdx.x * blockDim.x + threadIdx.x;
    if (i >= total) return;
    float wv = w[i];
    _Float16 hi = (_Float16)wv;
    _Float16 lo = (_Float16)(wv - (float)hi);
    whi[i] = hi;
    wlo[i] = lo;
}

// ---------------------------------------------------------------------------
// Async global->LDS copy (CDNA5 GLOBAL_LOAD_ASYNC_TO_LDS, ASYNCcnt-tracked).
// ldsoff = LDS byte address (low 32 bits of the generic shared pointer).
// ---------------------------------------------------------------------------
__device__ __forceinline__ void async16(uint32_t ldsoff, const _Float16* g)
{
    asm volatile("global_load_async_to_lds_b128 %0, %1, off"
                 :: "v"(ldsoff), "v"(g)
                 : "memory");
}

// ---------------------------------------------------------------------------
// Kernel 3: WMMA GEMM, triple-buffered async-to-LDS pipeline (depth 2,
// one barrier per K-step).
// y[m,n] = ts[m] * sum_k qx[m,k]*(whi[n,k]+wlo[n,k]) + b[n]
// Block tile 128x128, K-step 32, 8 waves; each wave 32(M) x 64(N).
// ---------------------------------------------------------------------------
#define BM 128
#define BN 128
#define BK 32
#define LDSPITCH (BK + 8)   // 80B rows: conflict-free 16B lane accesses
#define NBUF 3

__global__ __launch_bounds__(256) void wmma_gemm_kernel(
    const _Float16* __restrict__ qx,
    const _Float16* __restrict__ whi,
    const _Float16* __restrict__ wlo,
    const float* __restrict__ ts,
    const float* __restrict__ bias,
    float* __restrict__ y,
    int M, int N, int K)
{
    __shared__ _Float16 lsA [NBUF][BM][LDSPITCH];
    __shared__ _Float16 lsBh[NBUF][BN][LDSPITCH];
    __shared__ _Float16 lsBl[NBUF][BN][LDSPITCH];

    const int tid  = threadIdx.x;
    const int lane = tid & 31;
    const int wid  = tid >> 5;     // 0..7
    const int wm   = wid >> 1;     // 0..3  (M direction, 32 rows each)
    const int wn   = wid & 1;      // 0..1  (N direction, 64 cols each)

    const int bm = blockIdx.y * BM;
    const int bn = blockIdx.x * BN;

    const int half = lane >> 4;    // 0 or 1
    const int l16  = lane & 15;

    v8f acc[2][4];
    #pragma unroll
    for (int i = 0; i < 2; ++i)
        #pragma unroll
        for (int j = 0; j < 4; ++j)
            acc[i][j] = (v8f)0.0f;

    // staging: each thread issues 6 async b128 copies (16 f16 per tile)
    const int srow = tid >> 1;           // 0..127
    const int scol = (tid & 1) * 16;     // 0 or 16

    auto issue = [&](int buf, int k0) {
        const _Float16* gA = qx  + (size_t)(bm + srow) * K + k0 + scol;
        const _Float16* gH = whi + (size_t)(bn + srow) * K + k0 + scol;
        const _Float16* gL = wlo + (size_t)(bn + srow) * K + k0 + scol;
        uint32_t aA = (uint32_t)(uintptr_t)&lsA [buf][srow][scol];
        uint32_t aH = (uint32_t)(uintptr_t)&lsBh[buf][srow][scol];
        uint32_t aL = (uint32_t)(uintptr_t)&lsBl[buf][srow][scol];
        async16(aA,      gA);
        async16(aA + 16, gA + 8);
        async16(aH,      gH);
        async16(aH + 16, gH + 8);
        async16(aL,      gL);
        async16(aL + 16, gL + 8);
    };

    auto compute = [&](int buf) {
        // ---- A fragments (A 16x32 in this code's contiguous k order, same as B):
        // lanes 0-15: row=l16, K = kb..kb+15 contiguous, kb=half*16
        H16 afrag[2];
        #pragma unroll
        for (int i = 0; i < 2; ++i) {
            int row = wm * 32 + i * 16 + l16;
            int kb  = half * 16;
            afrag[i].h[0] = *(const v8h*)&lsA[buf][row][kb];
            afrag[i].h[1] = *(const v8h*)&lsA[buf][row][kb + 8];
        }
        // ---- B fragments (B 32x16 in this code's contiguous k order, same as A):
        // lane holds column n=l16, K = half*16 .. half*16+15 contiguous
        H16 bh[4], bl[4];
        #pragma unroll
        for (int j = 0; j < 4; ++j) {
            int col = wn * 64 + j * 16 + l16;
            int kb  = half * 16;
            bh[j].h[0] = *(const v8h*)&lsBh[buf][col][kb];
            bh[j].h[1] = *(const v8h*)&lsBh[buf][col][kb + 8];
            bl[j].h[0] = *(const v8h*)&lsBl[buf][col][kb];
            bl[j].h[1] = *(const v8h*)&lsBl[buf][col][kb + 8];
        }
        #pragma unroll
        for (int i = 0; i < 2; ++i) {
            #pragma unroll
            for (int j = 0; j < 4; ++j) {
                acc[i][j] = __builtin_amdgcn_wmma_f32_16x16x32_f16(
                    false, afrag[i].v, false, bh[j].v,
                    (short)0, acc[i][j], false, false);
                acc[i][j] = __builtin_amdgcn_wmma_f32_16x16x32_f16(
                    false, afrag[i].v, false, bl[j].v,
                    (short)0, acc[i][j], false, false);
            }
        }
    };

    const int nk = K / BK;   // 128 here; pipeline assumes nk >= 2

    // pipeline prologue: two stages in flight
    issue(0, 0);
    issue(1, BK);

    int cb = 0;   // buffer computed this iteration
    int ib = 2;   // buffer issued this iteration

    // steady state: branch-free body, one barrier per K-step
    for (int kt = 0; kt < nk - 2; ++kt) {
        // oldest 6 copies (buffer cb) complete; 6 stay in flight
        asm volatile("s_wait_asynccnt 0x6" ::: "memory");
        __syncthreads();   // cb visible to all; all waves done reading ib
        issue(ib, (kt + 2) * BK);
        compute(cb);
        cb = (cb == NBUF - 1) ? 0 : cb + 1;
        ib = (ib == NBUF - 1) ? 0 : ib + 1;
    }

    // epilogue step nk-2: one stage still in flight
    asm volatile("s_wait_asynccnt 0x6" ::: "memory");
    __syncthreads();
    compute(cb);
    cb = (cb == NBUF - 1) ? 0 : cb + 1;

    // epilogue step nk-1: drain
    asm volatile("s_wait_asynccnt 0x0" ::: "memory");
    __syncthreads();
    compute(cb);

    // Epilogue: C/D layout: VGPR r, lane -> m_local = r + half*8, n_local = l16
    #pragma unroll
    for (int i = 0; i < 2; ++i) {
        #pragma unroll
        for (int j = 0; j < 4; ++j) {
            int n = bn + wn * 64 + j * 16 + l16;
            float bv = bias[n];
            #pragma unroll
            for (int r = 0; r < 8; ++r) {
                int m = bm + wm * 32 + i * 16 + r + half * 8;
                y[(size_t)m * N + n] = ts[m] * acc[i][j][r] + bv;
            }
        }
    }
}

// ---------------------------------------------------------------------------
extern "C" void kernel_launch(void* const* d_in, const int* in_sizes, int n_in,
                              void* d_out, int out_size, void* d_ws, size_t ws_size,
                              hipStream_t stream)
{
    const float* x      = (const float*)d_in[0];   // (B,S,DIN)
    const float* weight = (const float*)d_in[1];   // (DOUT,DIN)
    const float* bias   = (const float*)d_in[2];   // (1,DOUT)
    const float* sscale = (const float*)d_in[3];   // (1,DIN)

    const int K = in_sizes[3];              // DIN  = 4096
    const int N = in_sizes[2];              // DOUT = 4096
    const int M = in_sizes[0] / K;          // B*S  = 8192

    // workspace layout
    char* ws = (char*)d_ws;
    size_t qxBytes = (size_t)M * K * sizeof(_Float16);
    size_t wBytes  = (size_t)N * K * sizeof(_Float16);
    _Float16* qx  = (_Float16*)ws;
    _Float16* whi = (_Float16*)(ws + qxBytes);
    _Float16* wlo = (_Float16*)(ws + qxBytes + wBytes);
    float*    ts  = (float*)(ws + qxBytes + 2 * wBytes);

    float* y = (float*)d_out;

    // 1) sparsify + quantize activations
    quant_sparsify_kernel<<<M, 256, 0, stream>>>(x, sscale, qx, ts, K);

    // 2) split weights into f16 hi/lo
    long long total = (long long)N * K;
    int blocks = (int)((total + 255) / 256);
    weight_split_kernel<<<blocks, 256, 0, stream>>>(weight, whi, wlo, total);

    // 3) WMMA GEMM with per-token scale + bias epilogue
    dim3 grid(N / BN, M / BM);
    wmma_gemm_kernel<<<grid, 256, 0, stream>>>(qx, whi, wlo, ts, bias, y, M, N, K);
}
